// PartialOT_89000312308415
// MI455X (gfx1250) — matrix operand order repaired and checked by reading, not compiled, not run on hardware
//
#include <hip/hip_runtime.h>
#include <cstdint>
#include <cstddef>

// ---------------------------------------------------------------------------
// PartialOT (SuperGlue-style Sinkhorn matching) for MI455X / gfx1250.
// B=4, D=64, m=n=4096, L=m+1=4097, 10 Sinkhorn iters.
//
//   1. pack_rows        : feat[b,c,p] f32 -> packed bf16 [b,p,c] + sq-norms
//   2. fill_bins        : bin_score borders of couplings (in d_out)
//   3. cdist_wmma       : dist = sqrt(max(x2+y2-2*f0·f1ᵀ,0)) via v_wmma bf16
//   4. 10x (row_pass ; col_pass_partial + col_combine) : log-Sinkhorn sweeps
//   5. finalize         : gamma = exp(c+u+v-norm) in-place + loss partials
//   6. final_reduce     : deterministic scalar losses -> d_out[0..1]
// ---------------------------------------------------------------------------

typedef __bf16 bf16;
typedef __attribute__((ext_vector_type(8)))  bf16  v8bf;
typedef __attribute__((ext_vector_type(16))) bf16  v16bf;
typedef __attribute__((ext_vector_type(8)))  float v8f;

constexpr int kB = 4;       // batch
constexpr int kD = 64;      // feature dim
constexpr int kM = 4096;    // m == n == h*w
constexpr int kL = 4097;    // m+1
constexpr int kChunk  = 256;                        // rows per col-pass chunk
constexpr int kNChunk = (kL + kChunk - 1) / kChunk; // 17
#define NORM_CONST (-9.01091334727928f)   // -log(8192)
#define LOGN_CONST ( 8.31776616671934f)   //  log(4096)

#define CIDX(b,i,j) ((((size_t)(b))*kL + (size_t)(i))*kL + (size_t)(j))

static __device__ __forceinline__ bf16 f32_to_bf16(float f) {
  uint32_t u = __builtin_bit_cast(uint32_t, f);
  uint32_t r = u + 0x7FFFu + ((u >> 16) & 1u);       // round-to-nearest-even
  uint16_t h = (uint16_t)(r >> 16);
  return __builtin_bit_cast(bf16, h);
}

static __device__ __forceinline__ void lse_acc(float& m, float& s, float x) {
  if (x > m) { s = s * __expf(m - x) + 1.0f; m = x; }
  else       { s += __expf(x - m); }
}
static __device__ __forceinline__ void lse_merge(float& m, float& s,
                                                 float m2, float s2) {
  float M = fmaxf(m, m2);
  s = s * __expf(m - M) + s2 * __expf(m2 - M);
  m = M;
}

// --------------------------------------------------------------------------
// 1. feat [b, c, p] (f32) -> packed [b, p, c] (bf16, 128B rows) + sq-norms.
// --------------------------------------------------------------------------
__global__ void pack_rows(const float* __restrict__ feat,
                          bf16* __restrict__ pf, float* __restrict__ sq) {
  int tid = blockIdx.x * blockDim.x + threadIdx.x;   // 0 .. B*M-1
  int b = tid >> 12;
  int p = tid & (kM - 1);
  const float* src = feat + (size_t)b * kD * kM + p;
  bf16* dst = pf + ((size_t)b * kM + p) * kD;
  float acc = 0.f;
#pragma unroll
  for (int g = 0; g < 8; ++g) {
    v8bf chunk;
#pragma unroll
    for (int e = 0; e < 8; ++e) {
      float val = src[(size_t)(g * 8 + e) * kM];
      acc = fmaf(val, val, acc);
      chunk[e] = f32_to_bf16(val);
    }
    *(v8bf*)(dst + g * 8) = chunk;
  }
  sq[tid] = acc;
}

// --------------------------------------------------------------------------
// 2. Borders of couplings: last column (i=0..4096) and last row (j=0..4095).
// --------------------------------------------------------------------------
__global__ void fill_bins(float* __restrict__ C, const float* __restrict__ binp) {
  const int per = kL + kM;                           // 8193 per batch
  int tid = blockIdx.x * blockDim.x + threadIdx.x;
  if (tid >= kB * per) return;
  float bin = binp[0];
  int b = tid / per, r = tid - b * per;
  if (r < kL) C[CIDX(b, r, kM)]      = bin;          // last column (+corner)
  else        C[CIDX(b, kM, r - kL)] = bin;          // last row
}

// --------------------------------------------------------------------------
// 3. cdist via V_WMMA_F32_16X16X32_BF16. One wave per 16x16 output tile;
//    8 waves per block cover a 16x128 strip. K=64 -> two WMMAs.
//    A frag (ISA 7.12.2): lanes 0-15 K{0-7,16-23}, lanes 16-31 K{8-15,24-31}.
//    B frag: lane n = column j0+n, lanes 0-15 K{0-7,16-23}, lanes 16-31 K{8-15,24-31}.
//    C/D: reg r -> M = r + 8*(lane>>4), N = lane&15.
// --------------------------------------------------------------------------
__global__ void cdist_wmma(const bf16* __restrict__ pf0,
                           const bf16* __restrict__ pf1,
                           const float* __restrict__ x2,
                           const float* __restrict__ y2,
                           float* __restrict__ C) {
  const int lane = threadIdx.x & 31;
  const int wave = threadIdx.x >> 5;
  const int b  = blockIdx.z;
  const int i0 = blockIdx.y * 16;
  const int j0 = blockIdx.x * 128 + wave * 16;
  const int half = lane >> 4;
  const int l15  = lane & 15;

  const bf16* arow = pf0 + ((size_t)b * kM + (i0 + l15)) * kD;
  const bf16* brow = pf1 + ((size_t)b * kM + (j0 + l15)) * kD;

  v8f acc = {0.f, 0.f, 0.f, 0.f, 0.f, 0.f, 0.f, 0.f};
#pragma unroll
  for (int kk = 0; kk < kD; kk += 32) {
    v8bf alo = *(const v8bf*)(arow + kk + 8 * half);
    v8bf ahi = *(const v8bf*)(arow + kk + 16 + 8 * half);
    v16bf a = __builtin_shufflevector(alo, ahi, 0, 1, 2, 3, 4, 5, 6, 7,
                                      8, 9, 10, 11, 12, 13, 14, 15);
    v8bf blo = *(const v8bf*)(brow + kk + 8 * half);
    v8bf bhi = *(const v8bf*)(brow + kk + 16 + 8 * half);
    v16bf bb = __builtin_shufflevector(blo, bhi, 0, 1, 2, 3, 4, 5, 6, 7,
                                       8, 9, 10, 11, 12, 13, 14, 15);
    acc = __builtin_amdgcn_wmma_f32_16x16x32_bf16(
        /*neg_a=*/false, a, /*neg_b=*/false, bb,
        /*c_mod=*/(short)0, acc, /*reuse_a=*/false, /*reuse_b=*/false);
  }

  const float yj = y2[(size_t)b * kM + j0 + l15];
#pragma unroll
  for (int r = 0; r < 8; ++r) {
    int i = i0 + r + 8 * half;
    float xi = x2[(size_t)b * kM + i];
    float d2 = fmaxf(xi + yj - 2.0f * acc[r], 0.0f);
    // raw v_sqrt_f32 (~1ulp): avoids the IEEE refinement VALU tail on the
    // 268MB-producing pass.
    C[CIDX(b, i, j0 + l15)] = __builtin_amdgcn_sqrtf(d2);
  }
}

// --------------------------------------------------------------------------
// 4a. u[b,i] = log_mu[i] - LSE_j( C[b,i,j] + v[b,j] ). Block per row, 256
//     threads stride contiguous columns; 4 independent online-LSE
//     accumulators per thread (ILP for the v_exp_f32 chain); LDS tree merge.
// --------------------------------------------------------------------------
__global__ void row_pass(const float* __restrict__ C,
                         const float* __restrict__ v,
                         float* __restrict__ u) {
  const int i = blockIdx.x, b = blockIdx.y, t = threadIdx.x;
  const float* row = C + CIDX(b, i, 0);
  const float* vb  = v + (size_t)b * kL;
  float m0 = -3.0e38f, s0 = 0.f, m1 = -3.0e38f, s1 = 0.f;
  float m2 = -3.0e38f, s2 = 0.f, m3 = -3.0e38f, s3 = 0.f;
  for (int j = t; j < kL; j += 1024) {
    lse_acc(m0, s0, row[j] + vb[j]);
    int j1 = j + 256, j2 = j + 512, j3 = j + 768;
    if (j1 < kL) lse_acc(m1, s1, row[j1] + vb[j1]);
    if (j2 < kL) lse_acc(m2, s2, row[j2] + vb[j2]);
    if (j3 < kL) lse_acc(m3, s3, row[j3] + vb[j3]);
  }
  lse_merge(m0, s0, m1, s1);
  lse_merge(m2, s2, m3, s3);
  lse_merge(m0, s0, m2, s2);
  __shared__ float sm[256], ss[256];
  sm[t] = m0; ss[t] = s0; __syncthreads();
  for (int off = 128; off > 0; off >>= 1) {
    if (t < off) lse_merge(sm[t], ss[t], sm[t + off], ss[t + off]);
    __syncthreads();
  }
  if (t == 0) {
    float lse    = sm[0] + __logf(ss[0]);
    float log_mu = (i < kM) ? NORM_CONST : (LOGN_CONST + NORM_CONST);
    u[(size_t)b * kL + i] = log_mu - lse;
  }
}

// --------------------------------------------------------------------------
// 4b-1. Column-pass partials: grid (colblocks=17, rowchunks=17, B). Each
//       block owns 256 columns x 256 rows; thread t = one column, rows
//       walked with 4-way-unrolled online LSE; u chunk staged in LDS.
//       Writes (max,sum) per (b, chunk, col). 1156 blocks -> saturates HBM.
// --------------------------------------------------------------------------
__global__ void col_pass_partial(const float* __restrict__ C,
                                 const float* __restrict__ u,
                                 float2* __restrict__ cpart) {
  const int t = threadIdx.x, b = blockIdx.z, chunk = blockIdx.y;
  const int j  = blockIdx.x * 256 + t;
  const int i0 = chunk * kChunk;
  __shared__ float sh_u[kChunk];
  if (i0 + t < kL) sh_u[t] = u[(size_t)b * kL + i0 + t];
  __syncthreads();
  const int lim = min(kChunk, kL - i0);
  float m0 = -3.0e38f, s0 = 0.f, m1 = -3.0e38f, s1 = 0.f;
  float m2 = -3.0e38f, s2 = 0.f, m3 = -3.0e38f, s3 = 0.f;
  if (j < kL) {
    const float* col = C + CIDX(b, i0, j);
    int ii = 0;
    for (; ii + 4 <= lim; ii += 4) {
      float x0 = col[(size_t)(ii + 0) * kL] + sh_u[ii + 0];
      float x1 = col[(size_t)(ii + 1) * kL] + sh_u[ii + 1];
      float x2v = col[(size_t)(ii + 2) * kL] + sh_u[ii + 2];
      float x3 = col[(size_t)(ii + 3) * kL] + sh_u[ii + 3];
      lse_acc(m0, s0, x0);
      lse_acc(m1, s1, x1);
      lse_acc(m2, s2, x2v);
      lse_acc(m3, s3, x3);
    }
    for (; ii < lim; ++ii)
      lse_acc(m0, s0, col[(size_t)ii * kL] + sh_u[ii]);
    lse_merge(m0, s0, m1, s1);
    lse_merge(m2, s2, m3, s3);
    lse_merge(m0, s0, m2, s2);
    cpart[((size_t)b * kNChunk + chunk) * kL + j] = make_float2(m0, s0);
  }
}

// --------------------------------------------------------------------------
// 4b-2. Combine chunk partials -> v[b,j]. Deterministic sequential merge.
// --------------------------------------------------------------------------
__global__ void col_combine(const float2* __restrict__ cpart,
                            float* __restrict__ v) {
  const int b = blockIdx.y;
  const int j = blockIdx.x * 256 + threadIdx.x;
  if (j >= kL) return;
  float m = -3.0e38f, s = 0.f;
#pragma unroll
  for (int c = 0; c < kNChunk; ++c) {
    float2 p = cpart[((size_t)b * kNChunk + c) * kL + j];
    lse_merge(m, s, p.x, p.y);
  }
  float lse    = m + __logf(s);
  float log_nu = (j < kM) ? NORM_CONST : (LOGN_CONST + NORM_CONST);
  v[(size_t)b * kL + j] = log_nu - lse;
}

// --------------------------------------------------------------------------
// 5. gamma = exp(c + u + v - norm) in place, fused with loss partials.
// --------------------------------------------------------------------------
__global__ void finalize(float* __restrict__ C,
                         const float* __restrict__ u,
                         const float* __restrict__ v,
                         float* __restrict__ partials) {
  const int i = blockIdx.x, b = blockIdx.y, t = threadIdx.x;
  float* row = C + CIDX(b, i, 0);
  const float* vb = v + (size_t)b * kL;
  const float  ui = u[(size_t)b * kL + i];
  const bool irow = (i < kM);
  float ml = 0.f, db1 = 0.f, db2 = 0.f;
  for (int j = t; j < kL; j += 256) {
    float c = row[j];
    float g = __expf(c + ui + vb[j] - NORM_CONST);  // Z - norm
    row[j] = g;
    if (irow && j < kM) ml += g * c;                // interior c == distance
    if (j == kM) db1 += g;
    if (irow) db2 += g;
  }
  __shared__ float s0[256], s1[256], s2[256];
  s0[t] = ml; s1[t] = db1; s2[t] = db2; __syncthreads();
  for (int off = 128; off > 0; off >>= 1) {
    if (t < off) { s0[t] += s0[t + off]; s1[t] += s1[t + off]; s2[t] += s2[t + off]; }
    __syncthreads();
  }
  if (t == 0) {
    size_t blk = (size_t)b * kL + i;
    partials[blk * 3 + 0] = s0[0];
    partials[blk * 3 + 1] = s1[0];
    partials[blk * 3 + 2] = s2[0];
  }
}

// --------------------------------------------------------------------------
// 6. Deterministic final reduction of block partials -> losses.
// --------------------------------------------------------------------------
__global__ void final_reduce(const float* __restrict__ partials,
                             float* __restrict__ out) {
  const int t = threadIdx.x;
  const int NP = kB * kL;
  float ml = 0.f, db1 = 0.f, db2 = 0.f;
  for (int idx = t; idx < NP; idx += 256) {
    ml  += partials[idx * 3 + 0];
    db1 += partials[idx * 3 + 1];
    db2 += partials[idx * 3 + 2];
  }
  __shared__ float s0[256], s1[256], s2[256];
  s0[t] = ml; s1[t] = db1; s2[t] = db2; __syncthreads();
  for (int off = 128; off > 0; off >>= 1) {
    if (t < off) { s0[t] += s0[t + off]; s1[t] += s1[t + off]; s2[t] += s2[t + off]; }
    __syncthreads();
  }
  if (t == 0) {
    out[0] = s0[0] / (float)(kB * kM);                              // matching
    out[1] = s1[0] / (float)(kB * kL)
           + s2[0] / ((float)kB * (float)kM * (float)kL);           // dustbin
  }
}

__global__ void zero_uv(float* __restrict__ u, float* __restrict__ v) {
  int tid = blockIdx.x * blockDim.x + threadIdx.x;
  if (tid < kB * kL) { u[tid] = 0.f; v[tid] = 0.f; }
}

// --------------------------------------------------------------------------
extern "C" void kernel_launch(void* const* d_in, const int* in_sizes, int n_in,
                              void* d_out, int out_size, void* d_ws, size_t ws_size,
                              hipStream_t stream) {
  (void)in_sizes; (void)n_in; (void)out_size; (void)ws_size;
  const float* feat0 = (const float*)d_in[0];
  const float* feat1 = (const float*)d_in[1];
  const float* bin   = (const float*)d_in[2];
  float* out = (float*)d_out;
  float* C   = out + 2;                        // gamma / couplings region

  char* ws = (char*)d_ws;
  bf16*   pf0      = (bf16*)(ws + 0);          // 2,097,152 B
  bf16*   pf1      = (bf16*)(ws + 2097152);    // 2,097,152 B
  float*  x2       = (float*)(ws + 4194304);   //    65,536 B
  float*  y2       = (float*)(ws + 4259840);   //    65,536 B
  float*  u        = (float*)(ws + 4325376);   //    65,552 B
  float*  v        = (float*)(ws + 4390928);   //    65,552 B
  float*  partials = (float*)(ws + 4456480);   //   196,656 B
  float2* cpart    = (float2*)(ws + 4653136);  // 2,228,768 B (4*17*4097*8)

  pack_rows<<<(kB * kM) / 256, 256, 0, stream>>>(feat0, pf0, x2);
  pack_rows<<<(kB * kM) / 256, 256, 0, stream>>>(feat1, pf1, y2);
  fill_bins<<<(kB * (kL + kM) + 255) / 256, 256, 0, stream>>>(C, bin);
  cdist_wmma<<<dim3(kM / 128, kM / 16, kB), 256, 0, stream>>>(pf0, pf1, x2, y2, C);
  zero_uv<<<(kB * kL + 255) / 256, 256, 0, stream>>>(u, v);
  for (int it = 0; it < 10; ++it) {
    row_pass<<<dim3(kL, kB), 256, 0, stream>>>(C, v, u);
    col_pass_partial<<<dim3((kL + 255) / 256, kNChunk, kB), 256, 0, stream>>>(C, u, cpart);
    col_combine<<<dim3((kL + 255) / 256, kB), 256, 0, stream>>>(cpart, v);
  }
  finalize<<<dim3(kL, kB), 256, 0, stream>>>(C, u, v, partials);
  final_reduce<<<1, 256, 0, stream>>>(partials, out);
}
